// CNN_TRX_83442624626847
// MI455X (gfx1250) — matrix operand order repaired and checked by reading, not compiled, not run on hardware
//
#include <hip/hip_runtime.h>
#include <math.h>

// ---------------- types ----------------
typedef __bf16 bf16_t;
typedef __bf16 v16bf __attribute__((ext_vector_type(16)));
typedef __bf16 v8bf  __attribute__((ext_vector_type(8)));
typedef float  v8f   __attribute__((ext_vector_type(8)));

#define SEQ_LEN 8
#define IN_DIM  2048
#define OUT_DIM 1152
#define FAN_IN  6144          // 3 * IN_DIM
#define NS      25
#define NQ      200
#define NT      56            // C(8,3)
#define NITEMS  225           // NS + NQ
#define MROWS   12600         // NITEMS * NT
#define SROWS   1400          // NS * NT
#define QROWS   11200         // NQ * NT

// C(8,3) frame triples in lexicographic (itertools.combinations) order
__constant__ int c_tup[56][3] = {
  {0,1,2},{0,1,3},{0,1,4},{0,1,5},{0,1,6},{0,1,7},
  {0,2,3},{0,2,4},{0,2,5},{0,2,6},{0,2,7},
  {0,3,4},{0,3,5},{0,3,6},{0,3,7},
  {0,4,5},{0,4,6},{0,4,7},
  {0,5,6},{0,5,7},
  {0,6,7},
  {1,2,3},{1,2,4},{1,2,5},{1,2,6},{1,2,7},
  {1,3,4},{1,3,5},{1,3,6},{1,3,7},
  {1,4,5},{1,4,6},{1,4,7},
  {1,5,6},{1,5,7},
  {1,6,7},
  {2,3,4},{2,3,5},{2,3,6},{2,3,7},
  {2,4,5},{2,4,6},{2,4,7},
  {2,5,6},{2,5,7},
  {2,6,7},
  {3,4,5},{3,4,6},{3,4,7},
  {3,5,6},{3,5,7},
  {3,6,7},
  {4,5,6},{4,5,7},
  {4,6,7},
  {5,6,7}
};

// ---------------- kernel 1: PE add + concat -> bf16 ----------------
__global__ void k_prep_x(const float* __restrict__ sup, const float* __restrict__ qry,
                         bf16_t* __restrict__ X) {
  int idx = blockIdx.x * 256 + threadIdx.x;
  if (idx >= NITEMS * SEQ_LEN * IN_DIM) return;
  int d    = idx & (IN_DIM - 1);
  int f    = (idx >> 11) & 7;
  int item = idx >> 14;
  float x = (item < NS) ? sup[idx] : qry[idx - NS * SEQ_LEN * IN_DIM];
  int i2 = d >> 1;
  // div = exp(-(2*i2) * ln(10000)/2048)
  float dv  = __expf(-(float)(2 * i2) * (9.210340371976184f / 2048.0f));
  float ang = (float)f * dv;
  float pe  = ((d & 1) ? __cosf(ang) : __sinf(ang)) * 0.1f;
  X[idx] = (bf16_t)(x + pe);
}

// ---------------- kernel 2: weight transpose [K][N] -> bf16 [N][K] ----------------
__global__ void k_wtrans(const float* __restrict__ kw, const float* __restrict__ vw,
                         bf16_t* __restrict__ wkt, bf16_t* __restrict__ wvt) {
  const float* src = blockIdx.z ? vw : kw;
  bf16_t*      dst = blockIdx.z ? wvt : wkt;
  __shared__ float tile[32][33];
  int n0 = blockIdx.x * 32;
  int k0 = blockIdx.y * 32;
  int tx = threadIdx.x, ty = threadIdx.y;  // (32, 8)
#pragma unroll
  for (int i = 0; i < 4; ++i)
    tile[ty + i * 8][tx] = src[(size_t)(k0 + ty + i * 8) * OUT_DIM + n0 + tx];
  __syncthreads();
#pragma unroll
  for (int i = 0; i < 4; ++i)
    dst[(size_t)(n0 + ty + i * 8) * FAN_IN + k0 + tx] = (bf16_t)tile[tx][ty + i * 8];
}

// ---------------- kernel 3: projection GEMM (tuple-gathered A, pipelined) ----------
// A: virtual [MROWS][FAN_IN] gathered from X via tuple table; B: WT [OUT_DIM][FAN_IN]
// mode 0: + bias -> bf16 KALL ; mode 1: + bias -> f32 VALL
__global__ void k_proj(const bf16_t* __restrict__ X, const bf16_t* __restrict__ WT,
                       const float* __restrict__ bias, bf16_t* __restrict__ koutb,
                       float* __restrict__ voutf, int mode) {
  const int lane    = threadIdx.x & 31;
  const int wave    = threadIdx.x >> 5;
  const int rowBase = blockIdx.x * 128 + wave * 16;
  const int colBase = blockIdx.y * 64;

  int r  = rowBase + (lane & 15);
  int rc = (r < MROWS - 1) ? r : (MROWS - 1);
  int item = rc / NT;
  int tup  = rc % NT;
  const bf16_t* xi = X + (size_t)item * (SEQ_LEN * IN_DIM);
  const int f0 = c_tup[tup][0], f1 = c_tup[tup][1], f2 = c_tup[tup][2];

  const int aoff = (lane & 16) ? 16 : 0;
  const int boff = (lane & 16) ? 16 : 0;
  // Segment-rebased A pointers: for kb in segment s, address = pa_s + kb (+8 for hi).
  // Both 8-half chunks of an iteration share the segment (kb%2048 <= 2016, offs <= 31).
  const bf16_t* pa0 = xi + (size_t)f0 * IN_DIM + aoff;
  const bf16_t* pa1 = xi + (size_t)f1 * IN_DIM + aoff - 2048;
  const bf16_t* pa2 = xi + (size_t)f2 * IN_DIM + aoff - 4096;

  const bf16_t* pB0 = WT + (size_t)(colBase +  0 + (lane & 15)) * FAN_IN + boff;
  const bf16_t* pB1 = WT + (size_t)(colBase + 16 + (lane & 15)) * FAN_IN + boff;
  const bf16_t* pB2 = WT + (size_t)(colBase + 32 + (lane & 15)) * FAN_IN + boff;
  const bf16_t* pB3 = WT + (size_t)(colBase + 48 + (lane & 15)) * FAN_IN + boff;

  v8f acc0 = {0.f,0.f,0.f,0.f,0.f,0.f,0.f,0.f};
  v8f acc1 = acc0, acc2 = acc0, acc3 = acc0;

  // ---- software pipeline: prologue loads for kb = 0 (segment 0) ----
  v8bf  loC = *(const v8bf*)(pa0);
  v8bf  hiC = *(const v8bf*)(pa0 + 8);
  v16bf b0C = *(const v16bf*)(pB0);
  v16bf b1C = *(const v16bf*)(pB1);
  v16bf b2C = *(const v16bf*)(pB2);
  v16bf b3C = *(const v16bf*)(pB3);

#pragma unroll 2
  for (int kb = 0; kb < FAN_IN; kb += 32) {
    int kn = kb + 32;
    kn = (kn < FAN_IN) ? kn : 0;           // last iter: dummy reload of kb=0
    int seg = kn >> 11;
    const bf16_t* pa = (seg == 0) ? pa0 : ((seg == 1) ? pa1 : pa2);
    // issue next-iteration loads before this iteration's WMMAs
    v8bf  loN = *(const v8bf*)(pa + kn);
    v8bf  hiN = *(const v8bf*)(pa + kn + 8);
    v16bf b0N = *(const v16bf*)(pB0 + kn);
    v16bf b1N = *(const v16bf*)(pB1 + kn);
    v16bf b2N = *(const v16bf*)(pB2 + kn);
    v16bf b3N = *(const v16bf*)(pB3 + kn);

    v16bf a = __builtin_shufflevector(loC, hiC, 0,1,2,3,4,5,6,7,8,9,10,11,12,13,14,15);
    acc0 = __builtin_amdgcn_wmma_f32_16x16x32_bf16(false, a, false, b0C, (short)0, acc0, false, false);
    acc1 = __builtin_amdgcn_wmma_f32_16x16x32_bf16(false, a, false, b1C, (short)0, acc1, false, false);
    acc2 = __builtin_amdgcn_wmma_f32_16x16x32_bf16(false, a, false, b2C, (short)0, acc2, false, false);
    acc3 = __builtin_amdgcn_wmma_f32_16x16x32_bf16(false, a, false, b3C, (short)0, acc3, false, false);

    loC = loN; hiC = hiN; b0C = b0N; b1C = b1N; b2C = b2N; b3C = b3N;
  }

  const int n0 = colBase + (lane & 15);
#pragma unroll
  for (int e = 0; e < 8; ++e) {
    int m = rowBase + e + ((lane & 16) ? 8 : 0);
    if (m >= MROWS) continue;
    size_t off = (size_t)m * OUT_DIM;
    if (mode == 0) {
      koutb[off + n0     ] = (bf16_t)(acc0[e] + bias[n0]);
      koutb[off + n0 + 16] = (bf16_t)(acc1[e] + bias[n0 + 16]);
      koutb[off + n0 + 32] = (bf16_t)(acc2[e] + bias[n0 + 32]);
      koutb[off + n0 + 48] = (bf16_t)(acc3[e] + bias[n0 + 48]);
    } else {
      voutf[off + n0     ] = acc0[e] + bias[n0];
      voutf[off + n0 + 16] = acc1[e] + bias[n0 + 16];
      voutf[off + n0 + 32] = acc2[e] + bias[n0 + 32];
      voutf[off + n0 + 48] = acc3[e] + bias[n0 + 48];
    }
  }
}

// ---------------- kernel 4: LayerNorm in-place on K (bf16) ----------------
__global__ void k_ln(bf16_t* __restrict__ K, const float* __restrict__ g,
                     const float* __restrict__ b) {
  int row = blockIdx.x;
  int t   = threadIdx.x;  // 128
  bf16_t* p = K + (size_t)row * OUT_DIM;
  float x[9];
  float s = 0.f;
#pragma unroll
  for (int i = 0; i < 9; ++i) { x[i] = (float)p[t + i * 128]; s += x[i]; }
  __shared__ float red[128];
  red[t] = s; __syncthreads();
  for (int o = 64; o > 0; o >>= 1) { if (t < o) red[t] += red[t + o]; __syncthreads(); }
  float mean = red[0] * (1.0f / OUT_DIM);
  __syncthreads();
  float sv = 0.f;
#pragma unroll
  for (int i = 0; i < 9; ++i) { float dd = x[i] - mean; sv += dd * dd; }
  red[t] = sv; __syncthreads();
  for (int o = 64; o > 0; o >>= 1) { if (t < o) red[t] += red[t + o]; __syncthreads(); }
  float rstd = rsqrtf(red[0] * (1.0f / OUT_DIM) + 1e-5f);
#pragma unroll
  for (int i = 0; i < 9; ++i) {
    int n = t + i * 128;
    p[n] = (bf16_t)((x[i] - mean) * rstd * g[n] + b[n]);
  }
}

// ---------------- kernel 5: scores GEMM  S[q*56+a][n*56+b]  (pipelined) ---------
__global__ void k_scores(const bf16_t* __restrict__ K, float* __restrict__ S) {
  const int lane    = threadIdx.x & 31;
  const int wave    = threadIdx.x >> 5;
  const int rowBase = blockIdx.x * 128 + wave * 16;
  const int colBase = blockIdx.y * 64;

  int m  = rowBase + (lane & 15);
  int mc = (m < QROWS - 1) ? m : (QROWS - 1);
  const int aoff = (lane & 16) ? 16 : 0;
  const int boff = (lane & 16) ? 16 : 0;
  const bf16_t* pA = K + (size_t)(SROWS + mc) * OUT_DIM + aoff;

  int j0 = colBase +  0 + (lane & 15); j0 = (j0 < SROWS - 1) ? j0 : (SROWS - 1);
  int j1 = colBase + 16 + (lane & 15); j1 = (j1 < SROWS - 1) ? j1 : (SROWS - 1);
  int j2 = colBase + 32 + (lane & 15); j2 = (j2 < SROWS - 1) ? j2 : (SROWS - 1);
  int j3 = colBase + 48 + (lane & 15); j3 = (j3 < SROWS - 1) ? j3 : (SROWS - 1);
  const bf16_t* pB0 = K + (size_t)j0 * OUT_DIM + boff;
  const bf16_t* pB1 = K + (size_t)j1 * OUT_DIM + boff;
  const bf16_t* pB2 = K + (size_t)j2 * OUT_DIM + boff;
  const bf16_t* pB3 = K + (size_t)j3 * OUT_DIM + boff;

  v8f acc0 = {0.f,0.f,0.f,0.f,0.f,0.f,0.f,0.f};
  v8f acc1 = acc0, acc2 = acc0, acc3 = acc0;

  // ---- software pipeline: prologue loads for kb = 0 ----
  v8bf  loC = *(const v8bf*)(pA);
  v8bf  hiC = *(const v8bf*)(pA + 8);
  v16bf b0C = *(const v16bf*)(pB0);
  v16bf b1C = *(const v16bf*)(pB1);
  v16bf b2C = *(const v16bf*)(pB2);
  v16bf b3C = *(const v16bf*)(pB3);

#pragma unroll 2
  for (int kb = 0; kb < OUT_DIM; kb += 32) {
    int kn = kb + 32;
    kn = (kn < OUT_DIM) ? kn : 0;          // last iter: dummy reload
    v8bf  loN = *(const v8bf*)(pA + kn);
    v8bf  hiN = *(const v8bf*)(pA + kn + 8);
    v16bf b0N = *(const v16bf*)(pB0 + kn);
    v16bf b1N = *(const v16bf*)(pB1 + kn);
    v16bf b2N = *(const v16bf*)(pB2 + kn);
    v16bf b3N = *(const v16bf*)(pB3 + kn);

    v16bf a = __builtin_shufflevector(loC, hiC, 0,1,2,3,4,5,6,7,8,9,10,11,12,13,14,15);
    acc0 = __builtin_amdgcn_wmma_f32_16x16x32_bf16(false, a, false, b0C, (short)0, acc0, false, false);
    acc1 = __builtin_amdgcn_wmma_f32_16x16x32_bf16(false, a, false, b1C, (short)0, acc1, false, false);
    acc2 = __builtin_amdgcn_wmma_f32_16x16x32_bf16(false, a, false, b2C, (short)0, acc2, false, false);
    acc3 = __builtin_amdgcn_wmma_f32_16x16x32_bf16(false, a, false, b3C, (short)0, acc3, false, false);

    loC = loN; hiC = hiN; b0C = b0N; b1C = b1N; b2C = b2N; b3C = b3N;
  }

  const float inv = 0.029462782549439483f;  // 1/sqrt(1152)
#pragma unroll
  for (int e = 0; e < 8; ++e) {
    int me = rowBase + e + ((lane & 16) ? 8 : 0);
    if (me >= QROWS) continue;
    size_t off = (size_t)me * SROWS;
    int n;
    n = colBase +  0 + (lane & 15); if (n < SROWS) S[off + n] = acc0[e] * inv;
    n = colBase + 16 + (lane & 15); if (n < SROWS) S[off + n] = acc1[e] * inv;
    n = colBase + 32 + (lane & 15); if (n < SROWS) S[off + n] = acc2[e] * inv;
    n = colBase + 48 + (lane & 15); if (n < SROWS) S[off + n] = acc3[e] * inv;
  }
}

// ---------------- kernel 6: per-class softmax in place ----------------
__global__ void k_softmax(float* __restrict__ S, const int* __restrict__ labels) {
  int row = blockIdx.x;   // q*56 + a
  int t   = threadIdx.x;  // 128
  float* p = S + (size_t)row * SROWS;
  __shared__ int   lab[NS];
  __shared__ float red[128];
  if (t < NS) lab[t] = labels[t];
  __syncthreads();

  float v[11]; int lb[11];
#pragma unroll
  for (int i = 0; i < 11; ++i) {
    int j = t + i * 128;
    if (j < SROWS) { v[i] = p[j]; lb[i] = lab[j / NT]; }
    else           { v[i] = 0.f;  lb[i] = -1; }
  }

  float mx[5];
#pragma unroll
  for (int c = 0; c < 5; ++c) mx[c] = -3.0e38f;
#pragma unroll
  for (int i = 0; i < 11; ++i) {
#pragma unroll
    for (int c = 0; c < 5; ++c) mx[c] = (lb[i] == c) ? fmaxf(mx[c], v[i]) : mx[c];
  }
#pragma unroll
  for (int c = 0; c < 5; ++c) {
    red[t] = mx[c]; __syncthreads();
    for (int o = 64; o > 0; o >>= 1) { if (t < o) red[t] = fmaxf(red[t], red[t + o]); __syncthreads(); }
    mx[c] = red[0]; __syncthreads();
  }

  float sm[5];
#pragma unroll
  for (int c = 0; c < 5; ++c) sm[c] = 0.f;
#pragma unroll
  for (int i = 0; i < 11; ++i) {
    float msel = mx[0];
#pragma unroll
    for (int c = 1; c < 5; ++c) msel = (lb[i] == c) ? mx[c] : msel;
    float e = __expf(v[i] - msel);
    e = (lb[i] < 0) ? 0.f : e;
    v[i] = e;
#pragma unroll
    for (int c = 0; c < 5; ++c) sm[c] += (lb[i] == c) ? e : 0.f;
  }
#pragma unroll
  for (int c = 0; c < 5; ++c) {
    red[t] = sm[c]; __syncthreads();
    for (int o = 64; o > 0; o >>= 1) { if (t < o) red[t] += red[t + o]; __syncthreads(); }
    sm[c] = red[0]; __syncthreads();
  }

#pragma unroll
  for (int i = 0; i < 11; ++i) {
    int j = t + i * 128;
    if (j < SROWS) {
      float ssel = sm[0];
#pragma unroll
      for (int c = 1; c < 5; ++c) ssel = (lb[i] == c) ? sm[c] : ssel;
      p[j] = v[i] / ssel;
    }
  }
}

// ---------------- kernel 7: prototype + squared distance ----------------
__global__ void k_proto(const float* __restrict__ S, const float* __restrict__ V,
                        const int* __restrict__ labels, float* __restrict__ out) {
  int c = blockIdx.x;   // class
  int q = blockIdx.y;   // query
  int t = threadIdx.x;  // 256
  __shared__ int   sup[NS];
  __shared__ int   ncs;
  __shared__ float attn[NT * NT];
  __shared__ float red[256];
  if (t == 0) {
    int cnt = 0;
    for (int n = 0; n < NS; ++n) if (labels[n] == c) sup[cnt++] = n;
    ncs = cnt;
  }
  __syncthreads();
  const int nc = ncs;
  const int dt = t & 31;   // d-lane
  const int at = t >> 5;   // a-group (0..7, 7 rows each)

  float dist = 0.f;
  for (int d0 = 0; d0 < OUT_DIM; d0 += 128) {
    float p0[7] = {}, p1[7] = {}, p2[7] = {}, p3[7] = {};
    for (int mi = 0; mi < nc; ++mi) {
      int sn = sup[mi];
      __syncthreads();
      for (int e = t; e < NT * NT; e += 256) {
        int a = e / NT, b = e - a * NT;
        attn[e] = S[(size_t)(q * NT + a) * SROWS + sn * NT + b];
      }
      __syncthreads();
      const float* vb = V + (size_t)sn * NT * OUT_DIM + d0 + dt;
      for (int b = 0; b < NT; ++b) {
        const float* vr = vb + (size_t)b * OUT_DIM;
        float v0 = vr[0], v1 = vr[32], v2 = vr[64], v3 = vr[96];
#pragma unroll
        for (int i = 0; i < 7; ++i) {
          float w = attn[(at * 7 + i) * NT + b];
          p0[i] += w * v0; p1[i] += w * v1; p2[i] += w * v2; p3[i] += w * v3;
        }
      }
    }
#pragma unroll
    for (int i = 0; i < 7; ++i) {
      int a = at * 7 + i;
      const float* qv = V + (size_t)(SROWS + q * NT + a) * OUT_DIM + d0 + dt;
      float e0 = qv[0] - p0[i], e1 = qv[32] - p1[i], e2 = qv[64] - p2[i], e3 = qv[96] - p3[i];
      dist += e0 * e0 + e1 * e1 + e2 * e2 + e3 * e3;
    }
  }
  red[t] = dist; __syncthreads();
  for (int o = 128; o > 0; o >>= 1) { if (t < o) red[t] += red[t + o]; __syncthreads(); }
  if (t == 0) out[q * 5 + c] = -red[0] / 56.0f;
}

// ---------------- launcher ----------------
extern "C" void kernel_launch(void* const* d_in, const int* in_sizes, int n_in,
                              void* d_out, int out_size, void* d_ws, size_t ws_size,
                              hipStream_t stream) {
  const float* sup    = (const float*)d_in[0];
  const int*   labels = (const int*)  d_in[1];
  const float* qry    = (const float*)d_in[2];
  const float* k_w    = (const float*)d_in[3];
  const float* k_b    = (const float*)d_in[4];
  const float* v_w    = (const float*)d_in[5];
  const float* v_b    = (const float*)d_in[6];
  const float* ln_g   = (const float*)d_in[7];
  const float* ln_b   = (const float*)d_in[8];
  float* out = (float*)d_out;

  char* ws = (char*)d_ws;
  bf16_t* X    = (bf16_t*)(ws);                 //   7,372,800 B
  bf16_t* WKT  = (bf16_t*)(ws + 7372800);       //  14,155,776 B
  bf16_t* WVT  = (bf16_t*)(ws + 21528576);      //  14,155,776 B
  bf16_t* KALL = (bf16_t*)(ws + 35684352);      //  29,030,400 B
  float*  VALL = (float*) (ws + 64714752);      //  58,060,800 B
  float*  SC   = (float*) (ws + 122775552);     //  62,720,000 B  (end ~177 MB)

  k_prep_x<<<dim3(14400), dim3(256), 0, stream>>>(sup, qry, X);
  k_wtrans<<<dim3(36, 192, 2), dim3(32, 8), 0, stream>>>(k_w, v_w, WKT, WVT);
  k_proj<<<dim3(99, 18), dim3(256), 0, stream>>>(X, WKT, k_b, KALL, VALL, 0);
  k_proj<<<dim3(99, 18), dim3(256), 0, stream>>>(X, WVT, v_b, KALL, VALL, 1);
  k_ln<<<dim3(12600), dim3(128), 0, stream>>>(KALL, ln_g, ln_b);
  k_scores<<<dim3(88, 22), dim3(256), 0, stream>>>(KALL, SC);
  k_softmax<<<dim3(11200), dim3(128), 0, stream>>>(SC, labels);
  k_proto<<<dim3(5, 200), dim3(256), 0, stream>>>(SC, VALL, labels, out);
}
